// SURFMNetLoss_26173530702517
// MI455X (gfx1250) — matrix operand order repaired and checked by reading, not compiled, not run on hardware
//
#include <hip/hip_runtime.h>

// ---------------------------------------------------------------------------
// SURFMNet loss on MI455X (gfx1250, wave32, WMMA + TDM).
// Dominant cost: M = E^T diag(f) E einsums (~159 GFLOP) -> bf16 WMMA 16x16x32
// with fp32 accumulation, fed by double-buffered Tensor-Data-Mover loads
// (tensor_load_to_lds, TENSORcnt-tracked, descriptor-side LDS row padding).
// Exact-fp32 128^3 matmuls (pre/bij/orth) -> f32 WMMA 16x16x4.
// Deterministic partial-sum reductions (no float atomics).
// ---------------------------------------------------------------------------

constexpr int Bb  = 4;
constexpr int Vv  = 5000;
constexpr int Ff  = 128;
constexpr int Kk  = 128;
constexpr int NDd = 121;          // int(128*0.95)
constexpr int VP  = 5024;         // V padded to multiple of 32 (157 chunks)
constexpr int NCH = VP / 32;

constexpr int NPRE   = 2 * Bb * NDd;  // 968 blocks for pre term
constexpr int NSMALL = Bb * 4;        // 16 blocks for bij/orth

typedef __attribute__((ext_vector_type(16))) __bf16 v16bf;
typedef __attribute__((ext_vector_type(8)))  float  v8f;
typedef __attribute__((ext_vector_type(2)))  float  v2f;
typedef __attribute__((ext_vector_type(4)))  unsigned int v4u;
typedef __attribute__((ext_vector_type(8)))  int    v8i;
typedef __attribute__((ext_vector_type(4)))  int    v4i;

// ---------------------------------------------------------------------------
// Deterministic block reduction (blockDim.x == 256)
// ---------------------------------------------------------------------------
__device__ __forceinline__ float block_reduce_256(float v, float* red) {
  int t = threadIdx.x;
  red[t] = v;
  __syncthreads();
  #pragma unroll
  for (int s = 128; s > 0; s >>= 1) {
    if (t < s) red[t] += red[t + s];
    __syncthreads();
  }
  float r = red[0];
  __syncthreads();
  return r;
}

// ---------------------------------------------------------------------------
// TDM descriptor: load a 32(v) x 128(k) bf16 tile (row stride VP elements)
// from global into LDS at lds_byte_off, padding each 64B row to 80B
// (pad_interval code 3 = 16 dwords, pad_amount code 3 = 4 dwords) to match
// the bank-conflict-padded Elds[][40] layout.
// ---------------------------------------------------------------------------
__device__ __forceinline__ void tdm_load_tile(const __bf16* gsrc,
                                              unsigned lds_byte_off) {
  unsigned long long ga = (unsigned long long)(size_t)gsrc;
  v4u g0;
  g0[0] = 1u;                                   // count=1, user mode
  g0[1] = lds_byte_off;                         // lds_addr
  g0[2] = (unsigned)(ga & 0xFFFFFFFFu);         // global_addr[31:0]
  g0[3] = (unsigned)((ga >> 32) & 0x01FFFFFFu)  // global_addr[56:32]
          | (2u << 30);                         // type=2 ("image")
  v8i g1;
  g1[0] = (1 << 16)      // data_size = 2 bytes
        | (1 << 20)      // pad_enable
        | (3 << 22)      // pad_interval: 16 dwords between pads
        | (3 << 25);     // pad_amount: 4 dwords
  g1[1] = (int)((VP & 0xFFFF) << 16);           // tensor_dim0[15:0]
  g1[2] = (int)(((VP >> 16) & 0xFFFF) | (Kk << 16));  // dim0 hi | dim1 lo
  g1[3] = (int)(32 << 16);                      // dim1 hi=0 | tile_dim0=32
  g1[4] = Kk;                                   // tile_dim1=128, tile_dim2=0
  g1[5] = VP;                                   // tensor_dim0_stride lo32
  g1[6] = 0;                                    // stride hi | dim1_stride lo
  g1[7] = 0;
  v4i z4 = {0, 0, 0, 0};
  v8i z8 = {0, 0, 0, 0, 0, 0, 0, 0};
  __builtin_amdgcn_tensor_load_to_lds(g0, g1, z4, z4, z8, 0);
}

// ---------------------------------------------------------------------------
// Prep: transpose + convert evecs -> Et[a][k][v] (bf16, zero-padded to VP)
// ---------------------------------------------------------------------------
__global__ void prep_evecs_kernel(const float* __restrict__ ev,
                                  __bf16* __restrict__ Et, int total) {
  for (int i = blockIdx.x * blockDim.x + threadIdx.x; i < total;
       i += gridDim.x * blockDim.x) {
    int v = i % VP;
    int r = (i / VP) % Kk;
    int a = i / (VP * Kk);
    float x = (v < Vv) ? ev[((size_t)a * Vv + v) * Kk + r] : 0.0f;
    Et[i] = (__bf16)x;
  }
}

// ---------------------------------------------------------------------------
// Prep: gather feat[:, :, descs] -> F[a][d][v] (f32, zero-padded to VP)
// ---------------------------------------------------------------------------
__global__ void prep_feat_kernel(const float* __restrict__ feat,
                                 const int* __restrict__ descs,
                                 float* __restrict__ Fg, int total) {
  for (int i = blockIdx.x * blockDim.x + threadIdx.x; i < total;
       i += gridDim.x * blockDim.x) {
    int v = i % VP;
    int d = (i / VP) % NDd;
    int a = i / (VP * NDd);
    Fg[i] = (v < Vv) ? feat[((size_t)a * Vv + v) * Ff + descs[d]] : 0.0f;
  }
}

// ---------------------------------------------------------------------------
// M kernel: M[a,d] = E^T diag(f_d) E via bf16 WMMA (fp32 accumulate).
// grid = 2*B*ND blocks, 256 threads (8 waves); wave w owns rows [16w,16w+16).
// Double-buffered TDM loads (wave 0 issues, s_wait_tensorcnt pipelines) stage
// 128(k) x 32(v) bf16 tiles; the f row is staged to LDS once at block start.
// ---------------------------------------------------------------------------
__global__ void m_kernel(const __bf16* __restrict__ Et1,
                         const __bf16* __restrict__ Et2,
                         const float* __restrict__ F1,
                         const float* __restrict__ F2,
                         float* __restrict__ M1, float* __restrict__ M2) {
  __shared__ __align__(16) __bf16 Elds[2][Kk][40];  // 2 x 10240 B
  __shared__ float flds[VP];                        // 20096 B

  int bi  = blockIdx.x;
  int s   = bi / (Bb * NDd);
  int rem = bi % (Bb * NDd);
  int a   = rem / NDd;
  int d   = rem % NDd;

  const __bf16* Et = s ? Et2 : Et1;
  const float*  Fg = s ? F2 : F1;
  float*        Mo = s ? M2 : M1;

  int tid  = threadIdx.x;
  int wave = tid >> 5;
  int lane = tid & 31;
  int m    = lane & 15;   // row within A tile / col within B/C tiles
  int h    = lane >> 4;   // lane half
  int k0   = wave * 16;

  const __bf16* Ebase = Et + (size_t)a * Kk * VP;
  const float*  fbase = Fg + ((size_t)a * NDd + d) * VP;

  unsigned ldsbuf0 = (unsigned)(size_t)(&Elds[0][0][0]);
  unsigned ldsbuf1 = (unsigned)(size_t)(&Elds[1][0][0]);

  // kick off chunk 0 into buffer 0 (wave 0 only; TDM ignores EXEC, gate by wave)
  if (wave == 0) tdm_load_tile(Ebase, ldsbuf0);

  // stage the whole f row once
  for (int i = tid; i < VP; i += 256) flds[i] = fbase[i];

  v8f acc[8];
  #pragma unroll
  for (int c = 0; c < 8; ++c)
    #pragma unroll
    for (int r = 0; r < 8; ++r) acc[c][r] = 0.0f;

  for (int ch = 0; ch < NCH; ++ch) {
    int buf = ch & 1;
    if (wave == 0) {
      if (ch + 1 < NCH) {
        tdm_load_tile(Ebase + (ch + 1) * 32,
                      (ch + 1) & 1 ? ldsbuf1 : ldsbuf0);
        __builtin_amdgcn_s_wait_tensorcnt(1);  // chunk ch resident (in-order)
      } else {
        __builtin_amdgcn_s_wait_tensorcnt(0);
      }
    }
    __syncthreads();  // tile (and, first iter, flds) visible to all waves

    int v0 = ch * 32;

    // A fragment: A[m][K] = f[K] * E[K][k0+m]  (16-bit A layout: K = h*8+e,
    // 16+h*8+e), scaled in fp32 then rounded to bf16.
    v16bf afrag;
    const __bf16* arow = &Elds[buf][k0 + m][0];
    #pragma unroll
    for (int e = 0; e < 8; ++e) {
      int kA = h * 8 + e;
      int kB = 16 + h * 8 + e;
      afrag[e]     = (__bf16)((float)arow[kA] * flds[v0 + kA]);
      afrag[8 + e] = (__bf16)((float)arow[kB] * flds[v0 + kB]);
    }

    // Preload all 8 B fragments, then chain the 8 WMMAs back-to-back.
    v16bf bfrag[8];
    #pragma unroll
    for (int c = 0; c < 8; ++c) {
      const __bf16* brow = &Elds[buf][c * 16 + m][0];
      #pragma unroll
      for (int e = 0; e < 16; ++e) bfrag[c][e] = brow[(e < 8) ? (h * 8 + e) : (8 + h * 8 + e)];
    }
    #pragma unroll
    for (int c = 0; c < 8; ++c)
      acc[c] = __builtin_amdgcn_wmma_f32_16x16x32_bf16(
          false, afrag, false, bfrag[c], (short)0, acc[c], false, false);

    __syncthreads();  // protect buf from being overwritten two chunks ahead
  }

  // C/D layout: VGPR r -> row = k0 + h*8 + r, col = c*16 + (lane%16)
  float* out = Mo + ((size_t)a * NDd + d) * Kk * Kk;
  #pragma unroll
  for (int c = 0; c < 8; ++c)
    #pragma unroll
    for (int r = 0; r < 8; ++r)
      out[(k0 + h * 8 + r) * Kk + c * 16 + m] = acc[c][r];
}

// ---------------------------------------------------------------------------
// pre kernel: per (a,d,s) compute sum((Cs@Ms - Mo@Cs)^2) with f32 WMMA.
// Dynamic LDS: Cl | Ms | Mo  (3 * 64KB = 192KB).
// ---------------------------------------------------------------------------
__global__ void pre_kernel(const float* __restrict__ C1,
                           const float* __restrict__ C2,
                           const float* __restrict__ M1,
                           const float* __restrict__ M2,
                           float* __restrict__ part) {
  extern __shared__ float lds[];
  float* Cl = lds;
  float* Ms = lds + Kk * Kk;
  float* Mo = lds + 2 * Kk * Kk;
  __shared__ float red[256];

  int bi  = blockIdx.x;
  int s   = bi / (Bb * NDd);
  int rem = bi % (Bb * NDd);
  int a   = rem / NDd;
  int d   = rem % NDd;

  const float* Cg  = (s ? C2 : C1) + (size_t)a * Kk * Kk;
  const float* Msg = (s ? M2 : M1) + ((size_t)a * NDd + d) * Kk * Kk;
  const float* Mog = (s ? M1 : M2) + ((size_t)a * NDd + d) * Kk * Kk;

  int tid = threadIdx.x;
  for (int i = tid; i < Kk * Kk; i += blockDim.x) {
    Cl[i] = Cg[i];
    Ms[i] = Msg[i];
    Mo[i] = Mog[i];
  }
  __syncthreads();

  int wave = tid >> 5, lane = tid & 31;
  int n = lane & 15, h = lane >> 4;
  int k0 = wave * 16;

  float local = 0.0f;
  #pragma unroll
  for (int c = 0; c < 8; ++c) {
    int l = c * 16 + n;
    v8f xacc, yacc;
    #pragma unroll
    for (int r = 0; r < 8; ++r) { xacc[r] = 0.0f; yacc[r] = 0.0f; }
    for (int kk = 0; kk < 32; ++kk) {
      int ki = kk * 4 + 2 * h;  // f32 A layout: lane half h -> K = 2h, 2h+1
      v2f aC, bM, aM, bC;
      aC.x = Cl[(k0 + n) * Kk + ki];
      aC.y = Cl[(k0 + n) * Kk + ki + 1];
      bM.x = Ms[ki * Kk + l];
      bM.y = Ms[(ki + 1) * Kk + l];
      xacc = __builtin_amdgcn_wmma_f32_16x16x4_f32(
          false, aC, false, bM, (short)0, xacc, false, false);
      aM.x = Mo[(k0 + n) * Kk + ki];
      aM.y = Mo[(k0 + n) * Kk + ki + 1];
      bC.x = Cl[ki * Kk + l];
      bC.y = Cl[(ki + 1) * Kk + l];
      yacc = __builtin_amdgcn_wmma_f32_16x16x4_f32(
          false, aM, false, bC, (short)0, yacc, false, false);
    }
    #pragma unroll
    for (int r = 0; r < 8; ++r) {
      float dd = xacc[r] - yacc[r];
      local += dd * dd;
    }
  }

  float total = block_reduce_256(local, red);
  if (tid == 0) part[bi] = total;
}

// ---------------------------------------------------------------------------
// bij/orth kernel: 16 blocks, block b=(a,t): t0:C1@C2 t1:C2@C1 t2:C1'@C2
// t3:C2'@C1; accumulate sum((P - I)^2). Dynamic LDS: A | B (128KB).
// ---------------------------------------------------------------------------
__global__ void small_kernel(const float* __restrict__ C1,
                             const float* __restrict__ C2,
                             float* __restrict__ part) {
  extern __shared__ float lds[];
  float* Al = lds;
  float* Bl = lds + Kk * Kk;
  __shared__ float red[256];

  int bi = blockIdx.x;
  int a  = bi >> 2;
  int t  = bi & 3;
  const float* Cb1 = C1 + (size_t)a * Kk * Kk;
  const float* Cb2 = C2 + (size_t)a * Kk * Kk;
  const float* As = (t == 0 || t == 2) ? Cb1 : Cb2;
  const float* Bs = (t == 0 || t == 2) ? Cb2 : Cb1;
  bool trA = (t >= 2);

  int tid = threadIdx.x;
  for (int i = tid; i < Kk * Kk; i += blockDim.x) {
    int r = i / Kk, c = i % Kk;
    Al[i] = trA ? As[c * Kk + r] : As[i];
    Bl[i] = Bs[i];
  }
  __syncthreads();

  int wave = tid >> 5, lane = tid & 31;
  int n = lane & 15, h = lane >> 4;
  int k0 = wave * 16;

  float local = 0.0f;
  #pragma unroll
  for (int c = 0; c < 8; ++c) {
    int l = c * 16 + n;
    v8f acc;
    #pragma unroll
    for (int r = 0; r < 8; ++r) acc[r] = 0.0f;
    for (int kk = 0; kk < 32; ++kk) {
      int ki = kk * 4 + 2 * h;
      v2f aa, bb;
      aa.x = Al[(k0 + n) * Kk + ki];
      aa.y = Al[(k0 + n) * Kk + ki + 1];
      bb.x = Bl[ki * Kk + l];
      bb.y = Bl[(ki + 1) * Kk + l];
      acc = __builtin_amdgcn_wmma_f32_16x16x4_f32(
          false, aa, false, bb, (short)0, acc, false, false);
    }
    #pragma unroll
    for (int r = 0; r < 8; ++r) {
      int row = k0 + h * 8 + r;
      float dd = acc[r] - ((row == l) ? 1.0f : 0.0f);
      local += dd * dd;
    }
  }

  float total = block_reduce_256(local, red);
  if (tid == 0) part[bi] = total;
}

// ---------------------------------------------------------------------------
// lap kernel: elementwise, one block, deterministic reduce.
// ---------------------------------------------------------------------------
__global__ void lap_kernel(const float* __restrict__ C1,
                           const float* __restrict__ C2,
                           const float* __restrict__ ev1,
                           const float* __restrict__ ev2,
                           float* __restrict__ part) {
  __shared__ float red[256];
  float local = 0.0f;
  for (int i = threadIdx.x; i < Bb * Kk * Kk; i += 256) {
    int a = i / (Kk * Kk);
    int r = (i / Kk) % Kk;
    int c = i % Kk;
    float c1 = C1[i], c2 = C2[i];
    float d1 = c1 * ev1[a * Kk + c] - ev2[a * Kk + r] * c1;
    float d2 = c2 * ev2[a * Kk + c] - ev1[a * Kk + r] * c2;
    local += d1 * d1 + d2 * d2;
  }
  float total = block_reduce_256(local, red);
  if (threadIdx.x == 0) part[0] = total;
}

// ---------------------------------------------------------------------------
// finalize: deterministic fixed-order sums -> out[4] = (bij, orth, lap, pre)
// ---------------------------------------------------------------------------
__global__ void finalize_kernel(const float* __restrict__ part_pre,
                                const float* __restrict__ part_small,
                                const float* __restrict__ part_lap,
                                float* __restrict__ out) {
  __shared__ float red[256];
  int tid = threadIdx.x;

  float lp = 0.0f;
  for (int i = tid; i < NPRE; i += 256) lp += part_pre[i];
  float Spre = block_reduce_256(lp, red);

  float lb = 0.0f, lo = 0.0f;
  if (tid < NSMALL) {
    float v = part_small[tid];
    if ((tid & 3) < 2) lb = v; else lo = v;
  }
  float Sbij  = block_reduce_256(lb, red);
  float Sorth = block_reduce_256(lo, red);

  if (tid == 0) {
    float Slap = part_lap[0];
    float nKK  = (float)(Bb * Kk * Kk);
    float nPre = (float)Bb * (float)NDd * (float)(Kk * Kk);
    out[0] = Sbij / nKK * 1000.0f;      // W_BIJ
    out[1] = Sorth / nKK * 1000.0f;     // W_ORTH
    out[2] = Slap / nKK * 1.0f;         // W_LAP
    out[3] = Spre / nPre * 100000.0f;   // W_PRE
  }
}

// ---------------------------------------------------------------------------
extern "C" void kernel_launch(void* const* d_in, const int* in_sizes, int n_in,
                              void* d_out, int out_size, void* d_ws,
                              size_t ws_size, hipStream_t stream) {
  const float* C1    = (const float*)d_in[0];
  const float* C2    = (const float*)d_in[1];
  const float* feat1 = (const float*)d_in[2];
  const float* feat2 = (const float*)d_in[3];
  const float* ev1   = (const float*)d_in[4];
  const float* ev2   = (const float*)d_in[5];
  const float* eval1 = (const float*)d_in[6];
  const float* eval2 = (const float*)d_in[7];
  const int*   descs = (const int*)d_in[8];

  char* ws = (char*)d_ws;
  size_t szEt = (size_t)Bb * Kk * VP * sizeof(__bf16);        // ~5.14 MB
  size_t szF  = (size_t)Bb * NDd * VP * sizeof(float);        // ~9.73 MB
  size_t szM  = (size_t)Bb * NDd * Kk * Kk * sizeof(float);   // ~31.7 MB

  __bf16* Et1 = (__bf16*)(ws);
  __bf16* Et2 = (__bf16*)(ws + szEt);
  float*  F1  = (float*)(ws + 2 * szEt);
  float*  F2  = (float*)(ws + 2 * szEt + szF);
  float*  M1  = (float*)(ws + 2 * szEt + 2 * szF);
  float*  M2  = (float*)(ws + 2 * szEt + 2 * szF + szM);
  float*  part       = (float*)(ws + 2 * szEt + 2 * szF + 2 * szM);
  float*  part_pre   = part;                  // NPRE floats
  float*  part_small = part + NPRE;           // NSMALL floats
  float*  part_lap   = part + NPRE + NSMALL;  // 1 float

  int totE = Bb * Kk * VP;
  prep_evecs_kernel<<<(totE + 255) / 256, 256, 0, stream>>>(ev1, Et1, totE);
  prep_evecs_kernel<<<(totE + 255) / 256, 256, 0, stream>>>(ev2, Et2, totE);
  int totF = Bb * NDd * VP;
  prep_feat_kernel<<<(totF + 255) / 256, 256, 0, stream>>>(feat1, descs, F1, totF);
  prep_feat_kernel<<<(totF + 255) / 256, 256, 0, stream>>>(feat2, descs, F2, totF);

  m_kernel<<<2 * Bb * NDd, 256, 0, stream>>>(Et1, Et2, F1, F2, M1, M2);

  pre_kernel<<<NPRE, 256, 3 * Kk * Kk * sizeof(float), stream>>>(C1, C2, M1,
                                                                 M2, part_pre);
  small_kernel<<<NSMALL, 256, 2 * Kk * Kk * sizeof(float), stream>>>(C1, C2,
                                                                     part_small);
  lap_kernel<<<1, 256, 0, stream>>>(C1, C2, eval1, eval2, part_lap);
  finalize_kernel<<<1, 256, 0, stream>>>(part_pre, part_small, part_lap,
                                         (float*)d_out);
}
